// LlamaAttention_26396869001230
// MI455X (gfx1250) — matrix operand order repaired and checked by reading, not compiled, not run on hardware
//
#include <hip/hip_runtime.h>

typedef __bf16 bf16;
typedef __attribute__((ext_vector_type(16))) __bf16 v16bf;
typedef __attribute__((ext_vector_type(8)))  float  v8f;
typedef __attribute__((ext_vector_type(4)))  unsigned int u32x4;
typedef __attribute__((ext_vector_type(8)))  int        i32x8;
typedef __attribute__((ext_vector_type(4)))  int        i32x4;

#define S_LEN   4096
#define HIDDEN  4096
#define NH      32
#define NKV     8
#define DHEAD   128
#define BLK     128
#define NBLK    (S_LEN / BLK)
#define LOCALNB 8
#define SINKNB  1

#if __has_builtin(__builtin_amdgcn_tensor_load_to_lds)
#define HAVE_TDM 1
#else
#define HAVE_TDM 0
#endif

union Frag { v16bf v; uint4 q[2]; };

__device__ __forceinline__ v8f wmma_bf16(const Frag& a, const Frag& b, v8f c) {
  return __builtin_amdgcn_wmma_f32_16x16x32_bf16(false, a.v, false, b.v,
                                                 (short)0, c, false, false);
}

// --------- fallback: per-lane async VMEM -> LDS copy (ASYNCcnt-tracked)
__device__ __forceinline__ void async_ld_b128(unsigned int lds_off, const void* gaddr) {
  asm volatile("global_load_async_to_lds_b128 %0, %1, off"
               :: "v"(lds_off), "v"(gaddr)
               : "memory");
}
__device__ __forceinline__ void wait_async0() {
  asm volatile("s_wait_asynccnt 0" ::: "memory");
}

#if HAVE_TDM
// --------- Tensor Data Mover: one instruction moves a whole 2D bf16 tile.
// D# per CDNA5 ISA. tensor dims == tile dims (no OOB),
// LDS padding replays our "+8 bf16 per row" pitch.
// This toolchain exposes the 6-arg builtin:
//   (u32x4 g0, i32x8 g1, i32x4 g2, i32x4 g3, i32x8 g4, i32 cpol)
__device__ __forceinline__ void tdm_load_2d(unsigned int lds_addr, const void* gptr,
                                            unsigned int tile_x, unsigned int tile_y,
                                            unsigned long long stride_elems,
                                            unsigned int pad_interval,  // 2^(v+1) DW
                                            unsigned int pad_dwords) {  // v+1 DW
  unsigned long long ga = (unsigned long long)(uintptr_t)gptr;
  u32x4 g0;
  g0.x = 1u;                                        // count=1 (valid descriptor)
  g0.y = lds_addr;                                  // lds_addr[31:0]
  g0.z = (unsigned int)ga;                          // global_addr[31:0]
  g0.w = ((unsigned int)(ga >> 32) & 0x01FFFFFFu)   // global_addr[56:32]
         | (2u << 30);                              // type=2 ("image")
  i32x8 g1;
  unsigned int w0 = (1u << 16);                     // data_size=1 (2 bytes)
  if (pad_dwords)
    w0 |= (1u << 20) | (pad_interval << 22) | ((pad_dwords - 1u) << 25);
  g1[0] = (int)w0;
  g1[1] = (int)((tile_x & 0xFFFFu) << 16);          // tensor_dim0[15:0] @ [31:16]
  g1[2] = (int)((tile_x >> 16) | ((tile_y & 0xFFFFu) << 16));   // dim0 hi | dim1 lo
  g1[3] = (int)((tile_y >> 16) | (tile_x << 16));   // dim1 hi | tile_dim0
  g1[4] = (int)(tile_y & 0xFFFFu);                  // tile_dim1 (tile_dim2 = 0)
  g1[5] = (int)(unsigned int)(stride_elems & 0xFFFFFFFFull);    // dim0_stride lo32
  g1[6] = (int)(unsigned int)((stride_elems >> 32) & 0xFFFFull); // dim0_stride hi16
  g1[7] = 0;
  i32x4 z4 = {0, 0, 0, 0};
  i32x8 z8 = {0, 0, 0, 0, 0, 0, 0, 0};
  __builtin_amdgcn_tensor_load_to_lds(g0, g1, z4, z4, z8, 0);
}
__device__ __forceinline__ void wait_tensor0() {
  __builtin_amdgcn_s_wait_tensorcnt(0);
}
#endif

// ---------------------------------------------------------------- f32 -> bf16
__global__ __launch_bounds__(256) void cvt_f32_bf16(const float* __restrict__ in,
                                                    bf16* __restrict__ out, int n) {
  int i = (blockIdx.x * 256 + threadIdx.x) * 4;
  if (i + 3 < n) {
    float4 f = *(const float4*)(in + i);
    out[i + 0] = (bf16)f.x;
    out[i + 1] = (bf16)f.y;
    out[i + 2] = (bf16)f.z;
    out[i + 3] = (bf16)f.w;
  }
}

// ------------------------------------------- f32 W[K][N] -> bf16 W^T[N][K]
__global__ __launch_bounds__(256) void cvt_transpose(const float* __restrict__ in,
                                                     bf16* __restrict__ out,
                                                     int K, int N) {
  __shared__ float tile[32][33];
  const int tx = threadIdx.x & 31;
  const int ty = threadIdx.x >> 5;           // 0..7
  const int n0 = blockIdx.x * 32;
  const int k0 = blockIdx.y * 32;
#pragma unroll
  for (int s = 0; s < 32; s += 8)
    tile[ty + s][tx] = in[(size_t)(k0 + ty + s) * N + n0 + tx];
  __syncthreads();
#pragma unroll
  for (int s = 0; s < 32; s += 8)
    out[(size_t)(n0 + ty + s) * K + k0 + tx] = (bf16)tile[tx][ty + s];
}

// ---------------------------------------------------------------- GEMM
// C[M,N] = A[M,K] * Bt[N,K]^T ; A,Bt bf16 row-major (Bt is pre-transposed B).
// EPI: 0 = f32 store, 1 = bf16 store, 2 = fused RoPE + bf16, 3 = bf16 transposed (Vt)
#define LDT 40   // 32 + 8 pad; rows stay 16B aligned (80B = 20 DW pitch)

template <int EPI, typename OutT>
__global__ __launch_bounds__(256) void gemm_bf16(const bf16* __restrict__ A,
                                                 const bf16* __restrict__ Bt,
                                                 OutT* __restrict__ C,
                                                 int M, int N, int K) {
  __shared__ bf16 As[2][128][LDT];
  __shared__ bf16 Bs[2][128][LDT];

  const int tid  = threadIdx.x;
  const int lane = tid & 31;
  const int wave = tid >> 5;        // 0..7 : 16 C-rows each
  const int l15  = lane & 15;
  const int half = lane >> 4;

  const int m0 = blockIdx.y * 128;
  const int n0 = blockIdx.x * 128;

  v8f acc[8] = {};

#if HAVE_TDM
  auto issue_tile = [&](int kb, int buf) {
    if (wave == 0) {
      // 128 rows x 32 bf16, row stride K; pad 4 DW every 16 DW -> 20 DW pitch
      tdm_load_2d((unsigned int)(uintptr_t)&As[buf][0][0],
                  A + (size_t)m0 * K + kb, 32, 128, (unsigned long long)K, 3, 4);
      tdm_load_2d((unsigned int)(uintptr_t)&Bs[buf][0][0],
                  Bt + (size_t)n0 * K + kb, 32, 128, (unsigned long long)K, 3, 4);
    }
  };
  auto wait_tiles = [&]() { if (wave == 0) wait_tensor0(); };
#else
  const int ldrow = tid >> 1;            // 0..127
  const int ldseg = (tid & 1) * 16;      // 0 / 16
  const bf16* arow = A  + (size_t)(m0 + ldrow) * K + ldseg;
  const bf16* brow = Bt + (size_t)(n0 + ldrow) * K + ldseg;
  auto issue_tile = [&](int kb, int buf) {
    unsigned int aoff = (unsigned int)(uintptr_t)&As[buf][ldrow][ldseg];
    unsigned int boff = (unsigned int)(uintptr_t)&Bs[buf][ldrow][ldseg];
    async_ld_b128(aoff,      arow + kb);
    async_ld_b128(aoff + 16, arow + kb + 8);
    async_ld_b128(boff,      brow + kb);
    async_ld_b128(boff + 16, brow + kb + 8);
  };
  auto wait_tiles = [&]() { wait_async0(); };
#endif

  issue_tile(0, 0);
  wait_tiles();
  __syncthreads();

  const int T = K >> 5;
  for (int t = 0; t < T; ++t) {
    const int buf = t & 1;
    if (t + 1 < T) issue_tile((t + 1) * 32, buf ^ 1);

    Frag af;
    af.q[0] = *(const uint4*)(&As[buf][wave * 16 + l15][8 * half]);
    af.q[1] = *(const uint4*)(&As[buf][wave * 16 + l15][16 + 8 * half]);
    Frag bfr[8];
#pragma unroll
    for (int n = 0; n < 8; ++n) {
      bfr[n].q[0] = *(const uint4*)(&Bs[buf][n * 16 + l15][8 * half]);
      bfr[n].q[1] = *(const uint4*)(&Bs[buf][n * 16 + l15][16 + 8 * half]);
    }
#pragma unroll
    for (int n = 0; n < 8; ++n)
      acc[n] = wmma_bf16(af, bfr[n], acc[n]);

    if (t + 1 < T) wait_tiles();
    __syncthreads();
  }

  // ---- epilogue: C frag element r -> row = r + 8*half, col = n*16 + l15
  if constexpr (EPI == 0) {
#pragma unroll
    for (int n = 0; n < 8; ++n)
#pragma unroll
      for (int r = 0; r < 8; ++r) {
        int gr = m0 + wave * 16 + r + 8 * half;
        int gc = n0 + n * 16 + l15;
        C[(size_t)gr * N + gc] = acc[n][r];
      }
  } else if constexpr (EPI == 1) {
#pragma unroll
    for (int n = 0; n < 8; ++n)
#pragma unroll
      for (int r = 0; r < 8; ++r) {
        int gr = m0 + wave * 16 + r + 8 * half;
        int gc = n0 + n * 16 + l15;
        C[(size_t)gr * N + gc] = (bf16)acc[n][r];
      }
  } else if constexpr (EPI == 3) {
    // V path: store transposed -> Vt[col][row]
#pragma unroll
    for (int n = 0; n < 8; ++n)
#pragma unroll
      for (int r = 0; r < 8; ++r) {
        int gr = m0 + wave * 16 + r + 8 * half;
        int gc = n0 + n * 16 + l15;
        C[(size_t)gc * S_LEN + gr] = (bf16)acc[n][r];
      }
  } else {
    // RoPE: 128-col tile == one head; pair (d, d+64) lives in frags (n, n+4)
#pragma unroll
    for (int n = 0; n < 4; ++n)
#pragma unroll
      for (int r = 0; r < 8; ++r) {
        int gr = m0 + wave * 16 + r + 8 * half;   // sequence position
        int d  = n * 16 + l15;                    // 0..63
        float freq = __powf(10000.0f, -(float)d * (1.0f / 64.0f));
        float ang  = (float)gr * freq;
        float sn, cs;
        __sincosf(ang, &sn, &cs);
        float x1 = acc[n][r];
        float x2 = acc[n + 4][r];
        size_t base = (size_t)gr * N + n0;
        C[base + d]      = (bf16)(x1 * cs - x2 * sn);
        C[base + d + 64] = (bf16)(x2 * cs + x1 * sn);
      }
  }
}

// ---------------------------------------------------------------- attention
#define LDK 136   // 128 + 8 pad, rows 16B aligned (272B = 68 DW pitch)

__global__ __launch_bounds__(256) void attn_kernel(const bf16* __restrict__ Q,
                                                   const bf16* __restrict__ Kg,
                                                   const bf16* __restrict__ Vt,
                                                   bf16* __restrict__ O) {
  extern __shared__ char smem[];
  bf16 (*Ks)[LDK] = (bf16(*)[LDK])(smem);                       // [key][d]
  bf16 (*Vs)[LDK] = (bf16(*)[LDK])(smem + 128 * LDK * 2);       // [d][key]
  bf16 (*Ps)[LDK] = (bf16(*)[LDK])(smem + 2 * 128 * LDK * 2);   // [q][key]

  const int tid  = threadIdx.x;
  const int lane = tid & 31;
  const int wave = tid >> 5;
  const int l15  = lane & 15;
  const int half = lane >> 4;
  const int i    = blockIdx.x;   // query block
  const int h    = blockIdx.y;   // head
  const int hkv  = h >> 2;       // NH/NKV = 4
  const float scale = 0.08838834764831845f;  // 1/sqrt(128)

  // Q A-frags for this wave's 16 rows, all 4 d-chunks, kept in registers
  Frag qf[4];
  {
    const int qrow = i * BLK + wave * 16 + l15;
    const bf16* qp = Q + (size_t)qrow * (NH * DHEAD) + h * DHEAD;
#pragma unroll
    for (int kc = 0; kc < 4; ++kc) {
      qf[kc].q[0] = *(const uint4*)(qp + kc * 32 + 8 * half);
      qf[kc].q[1] = *(const uint4*)(qp + kc * 32 + 16 + 8 * half);
    }
  }

  v8f oacc[8] = {};
  float mrow[8], lrow[8];
#pragma unroll
  for (int r = 0; r < 8; ++r) { mrow[r] = -1e30f; lrow[r] = 0.0f; }

  const int lo = (i - (LOCALNB - 1) > SINKNB) ? (i - (LOCALNB - 1)) : SINKNB;

#if !HAVE_TDM
  const int srow = tid >> 1;
  const int sseg = (tid & 1) * 64;
  const unsigned int koff = (unsigned int)(uintptr_t)&Ks[srow][sseg];
  const unsigned int voff = (unsigned int)(uintptr_t)&Vs[srow][sseg];
#endif

  for (int jb = lo - 1; jb <= i; ++jb) {
    const int j = (jb == lo - 1) ? 0 : jb;   // first pass = sink block 0

    // stage K[key][d] and Vt[d][key] blocks into LDS
#if HAVE_TDM
    if (wave == 0) {
      tdm_load_2d((unsigned int)(uintptr_t)&Ks[0][0],
                  Kg + (size_t)(j * BLK) * (NKV * DHEAD) + hkv * DHEAD,
                  128, 128, (unsigned long long)(NKV * DHEAD), 5, 4);
      tdm_load_2d((unsigned int)(uintptr_t)&Vs[0][0],
                  Vt + (size_t)(hkv * DHEAD) * S_LEN + j * BLK,
                  128, 128, (unsigned long long)S_LEN, 5, 4);
      wait_tensor0();
    }
#else
    {
      const bf16* kp = Kg + (size_t)(j * BLK + srow) * (NKV * DHEAD) + hkv * DHEAD + sseg;
      const bf16* vp = Vt + (size_t)(hkv * DHEAD + srow) * S_LEN + j * BLK + sseg;
#pragma unroll
      for (int c = 0; c < 8; ++c) {
        async_ld_b128(koff + c * 16, kp + c * 8);
        async_ld_b128(voff + c * 16, vp + c * 8);
      }
    }
    wait_async0();
#endif
    __syncthreads();

    // S = Q * K^T  (wave: 16 q-rows x 128 keys)
    v8f sacc[8] = {};
#pragma unroll
    for (int kc = 0; kc < 4; ++kc) {
      Frag kf[8];
#pragma unroll
      for (int n = 0; n < 8; ++n) {
        kf[n].q[0] = *(const uint4*)(&Ks[n * 16 + l15][kc * 32 + 8 * half]);
        kf[n].q[1] = *(const uint4*)(&Ks[n * 16 + l15][kc * 32 + 16 + 8 * half]);
      }
#pragma unroll
      for (int n = 0; n < 8; ++n)
        sacc[n] = wmma_bf16(qf[kc], kf[n], sacc[n]);
    }

    // mask + online softmax (rows stay inside the wave; reduce over 16 lanes)
    const int qbase = i * BLK + wave * 16;
    float pm[8];
#pragma unroll
    for (int r = 0; r < 8; ++r) {
      const int qg = qbase + r + 8 * half;
      float mx = -1e30f;
#pragma unroll
      for (int n = 0; n < 8; ++n) {
        int kg = j * BLK + n * 16 + l15;
        float v = sacc[n][r] * scale;
        if (kg > qg) v = -1e30f;
        sacc[n][r] = v;
        mx = fmaxf(mx, v);
      }
#pragma unroll
      for (int off = 8; off >= 1; off >>= 1)
        mx = fmaxf(mx, __shfl_xor(mx, off, 32));
      pm[r] = mx;
    }
#pragma unroll
    for (int r = 0; r < 8; ++r) {
      float mn = fmaxf(mrow[r], pm[r]);
      float fr = __expf(mrow[r] - mn);
      mrow[r] = mn;
      float sum = 0.0f;
#pragma unroll
      for (int n = 0; n < 8; ++n) {
        float p = __expf(sacc[n][r] - mn);
        sum += p;
        Ps[wave * 16 + r + 8 * half][n * 16 + l15] = (bf16)p;
      }
#pragma unroll
      for (int off = 8; off >= 1; off >>= 1)
        sum += __shfl_xor(sum, off, 32);
      lrow[r] = lrow[r] * fr + sum;
#pragma unroll
      for (int n = 0; n < 8; ++n) oacc[n][r] *= fr;
    }
    __syncthreads();

    // O += P * V
#pragma unroll
    for (int kc = 0; kc < 4; ++kc) {
      Frag pa;
      pa.q[0] = *(const uint4*)(&Ps[wave * 16 + l15][kc * 32 + 8 * half]);
      pa.q[1] = *(const uint4*)(&Ps[wave * 16 + l15][kc * 32 + 16 + 8 * half]);
      Frag vf[8];
#pragma unroll
      for (int n = 0; n < 8; ++n) {
        vf[n].q[0] = *(const uint4*)(&Vs[n * 16 + l15][kc * 32 + 8 * half]);
        vf[n].q[1] = *(const uint4*)(&Vs[n * 16 + l15][kc * 32 + 16 + 8 * half]);
      }
#pragma unroll
      for (int n = 0; n < 8; ++n)
        oacc[n] = wmma_bf16(pa, vf[n], oacc[n]);
    }
    __syncthreads();
  }

  // normalize + store
#pragma unroll
  for (int r = 0; r < 8; ++r) {
    const int qg = i * BLK + wave * 16 + r + 8 * half;
    float inv = 1.0f / lrow[r];
    bf16* op = O + (size_t)qg * (NH * DHEAD) + h * DHEAD;
#pragma unroll
    for (int n = 0; n < 8; ++n)
      op[n * 16 + l15] = (bf16)(oacc[n][r] * inv);
  }
}

// ---------------------------------------------------------------- launch
extern "C" void kernel_launch(void* const* d_in, const int* in_sizes, int n_in,
                              void* d_out, int out_size, void* d_ws, size_t ws_size,
                              hipStream_t stream) {
  const float* hs = (const float*)d_in[0];
  const float* wq = (const float*)d_in[1];
  const float* wk = (const float*)d_in[2];
  const float* wv = (const float*)d_in[3];
  const float* wo = (const float*)d_in[4];
  float* out = (float*)d_out;

  char* ws = (char*)d_ws;
  size_t off = 0;
  auto alloc_bf = [&](size_t elems) {
    bf16* p = (bf16*)(ws + off);
    off += elems * sizeof(bf16);
    return p;
  };
  const size_t NSQ  = (size_t)S_LEN * HIDDEN;       // 16.7M
  const size_t NKVE = (size_t)S_LEN * NKV * DHEAD;  // 4.2M

  bf16* Xb   = alloc_bf(NSQ);    // hidden, bf16
  bf16* Wqt  = alloc_bf(NSQ);    // wq^T  [N][K]
  bf16* Wkt  = alloc_bf(NKVE);   // wk^T
  bf16* Wvt  = alloc_bf(NKVE);   // wv^T
  bf16* Wot  = alloc_bf(NSQ);    // wo^T
  bf16* Qb   = alloc_bf(NSQ);    // Q after RoPE  [s][h*128+d]
  bf16* Kb   = alloc_bf(NKVE);   // K after RoPE  [s][hkv*128+d]
  bf16* Vtb  = alloc_bf(NKVE);   // V transposed  [hkv*128+d][s]
  bf16* Ab   = alloc_bf(NSQ);    // attention output

  // conversions / weight transposes
  cvt_f32_bf16<<<(int)(NSQ / 1024), 256, 0, stream>>>(hs, Xb, (int)NSQ);
  cvt_transpose<<<dim3(HIDDEN / 32, HIDDEN / 32), 256, 0, stream>>>(wq, Wqt, HIDDEN, HIDDEN);
  cvt_transpose<<<dim3((NKV * DHEAD) / 32, HIDDEN / 32), 256, 0, stream>>>(wk, Wkt, HIDDEN, NKV * DHEAD);
  cvt_transpose<<<dim3((NKV * DHEAD) / 32, HIDDEN / 32), 256, 0, stream>>>(wv, Wvt, HIDDEN, NKV * DHEAD);
  cvt_transpose<<<dim3(HIDDEN / 32, HIDDEN / 32), 256, 0, stream>>>(wo, Wot, HIDDEN, HIDDEN);

  dim3 blk(256);
  dim3 gQ(HIDDEN / 128, S_LEN / 128);
  dim3 gKV((NKV * DHEAD) / 128, S_LEN / 128);

  // projections (RoPE fused into Q/K epilogues; V stored transposed)
  gemm_bf16<2, bf16><<<gQ, blk, 0, stream>>>(Xb, Wqt, Qb, S_LEN, HIDDEN, HIDDEN);
  gemm_bf16<2, bf16><<<gKV, blk, 0, stream>>>(Xb, Wkt, Kb, S_LEN, NKV * DHEAD, HIDDEN);
  gemm_bf16<3, bf16><<<gKV, blk, 0, stream>>>(Xb, Wvt, Vtb, S_LEN, NKV * DHEAD, HIDDEN);

  // attention (dynamic LDS: 3 tiles of 128x136 bf16 ~ 102KB of the 320KB WGP pool)
  const int smem_bytes = 3 * 128 * LDK * (int)sizeof(bf16);
  (void)hipFuncSetAttribute((const void*)attn_kernel,
                            hipFuncAttributeMaxDynamicSharedMemorySize, smem_bytes);
  attn_kernel<<<dim3(NBLK, NH), blk, smem_bytes, stream>>>(Qb, Kb, Vtb, Ab);

  // output projection, f32 result
  gemm_bf16<0, float><<<gQ, blk, 0, stream>>>(Ab, Wot, out, S_LEN, HIDDEN, HIDDEN);
}
